// Recogniton_10849087389969
// MI455X (gfx1250) — matrix operand order repaired and checked by reading, not compiled, not run on hardware
//
#include <hip/hip_runtime.h>
#include <hip/hip_bf16.h>
#include <math.h>

// ---------------- types ----------------
typedef __bf16 bf16_t;
typedef __attribute__((ext_vector_type(16))) __bf16 v16bf;
typedef __attribute__((ext_vector_type(8)))  __bf16 v8bf;
typedef __attribute__((ext_vector_type(8)))  float  v8f;

#define BB 64
#define FF 512
#define HH 512
#define EE 512
#define MH 6
#define MW 40
#define HW 240          // MH*MW
#define TT 32           // decode steps
#define V2 96
#define ENDTOK 95
#define MROWS (BB*HW)   // 15360

// ---------------- WMMA fragment helpers (wave32, bf16 16x16x32) ----------------
// A 16x32 frag from row-major [M][K]: lane l -> row (l&15); elements 0..15 =
// K[g*16 .. g*16+15], g = l>>4.  (one 32B contiguous load)
__device__ __forceinline__ v16bf load_a_frag(const bf16_t* __restrict__ row, int k0) {
    int g = (threadIdx.x & 31) >> 4;
    union { v16bf v; v8bf h[2]; } u;
    u.v = *(const v16bf*)(row + k0 + g * 16);
    return u.v;
}
// B 32x16 frag from row-major weight [N][K]: lane l -> col n=(l&15); elements 0..15 =
// K[g*16 .. g*16+15] of row n.  (one 32B contiguous load)
__device__ __forceinline__ v16bf load_b_frag(const bf16_t* __restrict__ W, int ldk,
                                             int nbase, int k0) {
    int l = threadIdx.x & 31;
    const bf16_t* p = W + (size_t)(nbase + (l & 15)) * ldk + k0 + ((l >> 4) << 4);
    return *(const v16bf*)p;
}
__device__ __forceinline__ v8f wmma_bf16(v16bf a, v16bf b, v8f c) {
    return __builtin_amdgcn_wmma_f32_16x16x32_bf16(false, a, false, b, (short)0, c,
                                                   false, false);
}
__device__ __forceinline__ float sigf(float x) { return 1.f / (1.f + __expf(-x)); }

// ---------------- utility kernels ----------------
__global__ void fillz_kernel(unsigned int* p, int nwords) {
    int i = blockIdx.x * blockDim.x + threadIdx.x;
    if (i < nwords) p[i] = 0u;
}
__global__ void f2bf_kernel(const float* __restrict__ s, bf16_t* __restrict__ d, int n) {
    int i = blockIdx.x * blockDim.x + threadIdx.x;
    if (i < n) d[i] = (bf16_t)s[i];
}
// att_conv_w [E][F][3][3] -> wtap [tap][E][F] bf16
__global__ void convw_reorg_kernel(const float* __restrict__ w, bf16_t* __restrict__ wtap) {
    int i = blockIdx.x * blockDim.x + threadIdx.x;
    if (i >= 9 * EE * FF) return;
    int f = i % FF, e = (i / FF) % EE, tap = i / (FF * EE);
    wtap[i] = (bf16_t)w[((size_t)e * FF + f) * 9 + tap];
}
// conv_f [B][F][6][40] -> xseq bf16 [W][B][F] = max over height
__global__ void maxpool_kernel(const float* __restrict__ cf, bf16_t* __restrict__ xseq) {
    int i = blockIdx.x * blockDim.x + threadIdx.x;
    if (i >= MW * BB * FF) return;
    int f = i % FF, b = (i / FF) % BB, w = i / (FF * BB);
    const float* p = cf + ((size_t)(b * FF + f) * MH) * MW + w;
    float m = p[0];
    #pragma unroll
    for (int y = 1; y < MH; ++y) m = fmaxf(m, p[y * MW]);
    xseq[((size_t)w * BB + b) * FF + f] = (bf16_t)m;
}
// conv_f [B][F][6][40] -> cfpad bf16 [B][8][42][F] (zero borders pre-filled)
__global__ void padfill_kernel(const float* __restrict__ cf, bf16_t* __restrict__ pad) {
    int i = blockIdx.x * blockDim.x + threadIdx.x;
    if (i >= BB * MH * MW * FF) return;
    int f = i % FF, x = (i / FF) % MW, y = (i / (FF * MW)) % MH, b = i / (FF * MW * MH);
    pad[(((size_t)(b * 8 + y + 1)) * 42 + (x + 1)) * FF + f] =
        (bf16_t)cf[(((size_t)(b * FF + f)) * MH + y) * MW + x];
}
// xs[t][b][:] = embed(START or target[b][t-2]) for t>=1
__global__ void xsembed_kernel(const float* __restrict__ embed, const int* __restrict__ tgt,
                               bf16_t* __restrict__ xs) {
    int i = blockIdx.x * blockDim.x + threadIdx.x;
    if (i >= (TT - 1) * BB * HH) return;
    int f = i % HH, b = (i / HH) % BB, t = i / (HH * BB) + 1;
    int tok = (t == 1) ? ENDTOK : tgt[b * 30 + t - 2];
    xs[((size_t)t * BB + b) * HH + f] = (bf16_t)embed[(size_t)tok * HH + f];
}

// ---------------- fused LSTM step: gates GEMM (WMMA) + cell update ----------------
// grid 32 blocks x 128 thr = 128 waves; wave -> (mt in 0..3, nt in 0..31)
__global__ void lstm_step_kernel(const bf16_t* __restrict__ x, const bf16_t* __restrict__ hprev,
                                 const bf16_t* __restrict__ wih, const bf16_t* __restrict__ whh,
                                 const float* __restrict__ bih, const float* __restrict__ bhh,
                                 float* __restrict__ c, bf16_t* __restrict__ hout) {
    int wid = blockIdx.x * (blockDim.x >> 5) + (threadIdx.x >> 5);
    int mt = wid & 3, nt = wid >> 2;
    int l = threadIdx.x & 31, ln = l & 15, lg = l >> 4;
    v8f acc[4];
    #pragma unroll
    for (int g = 0; g < 4; ++g) {
        float bv = bih[g * HH + nt * 16 + ln] + bhh[g * HH + nt * 16 + ln];
        #pragma unroll
        for (int v = 0; v < 8; ++v) acc[g][v] = bv;
    }
    const bf16_t* xrow = x + (size_t)(mt * 16 + ln) * HH;
    const bf16_t* hrow = hprev + (size_t)(mt * 16 + ln) * HH;
    for (int kt = 0; kt < HH / 32; ++kt) {
        v16bf a = load_a_frag(xrow, kt * 32);
        #pragma unroll
        for (int g = 0; g < 4; ++g)
            acc[g] = wmma_bf16(a, load_b_frag(wih, HH, g * HH + nt * 16, kt * 32), acc[g]);
    }
    for (int kt = 0; kt < HH / 32; ++kt) {
        v16bf a = load_a_frag(hrow, kt * 32);
        #pragma unroll
        for (int g = 0; g < 4; ++g)
            acc[g] = wmma_bf16(a, load_b_frag(whh, HH, g * HH + nt * 16, kt * 32), acc[g]);
    }
    #pragma unroll
    for (int v = 0; v < 8; ++v) {
        int m = mt * 16 + lg * 8 + v, n = nt * 16 + ln;
        size_t idx = (size_t)m * HH + n;
        float ig = sigf(acc[0][v]), fg = sigf(acc[1][v]);
        float gg = tanhf(acc[2][v]), og = sigf(acc[3][v]);
        float cn = fg * c[idx] + ig * gg;
        c[idx] = cn;
        hout[idx] = (bf16_t)(og * tanhf(cn));
    }
}

// ---------------- generic M=64 GEMM: out = A1@W1^T (+A2@W2^T) + bias ----------------
__global__ void gemm_bias_kernel(const bf16_t* __restrict__ A1, const bf16_t* __restrict__ W1,
                                 int K1, int ldw1, const bf16_t* __restrict__ A2,
                                 const bf16_t* __restrict__ W2, int K2, int ldw2,
                                 const float* __restrict__ bias, float* __restrict__ out,
                                 int N, int ntiles) {
    int wid = blockIdx.x * (blockDim.x >> 5) + (threadIdx.x >> 5);
    if (wid >= 4 * ntiles) return;
    int mt = wid & 3, nt = wid >> 2;
    int l = threadIdx.x & 31, ln = l & 15, lg = l >> 4;
    v8f acc;
    float bv = bias[nt * 16 + ln];
    #pragma unroll
    for (int v = 0; v < 8; ++v) acc[v] = bv;
    const bf16_t* a1row = A1 + (size_t)(mt * 16 + ln) * K1;
    for (int kt = 0; kt < K1 / 32; ++kt)
        acc = wmma_bf16(load_a_frag(a1row, kt * 32),
                        load_b_frag(W1, ldw1, nt * 16, kt * 32), acc);
    if (A2) {
        const bf16_t* a2row = A2 + (size_t)(mt * 16 + ln) * K2;
        for (int kt = 0; kt < K2 / 32; ++kt)
            acc = wmma_bf16(load_a_frag(a2row, kt * 32),
                            load_b_frag(W2, ldw2, nt * 16, kt * 32), acc);
    }
    #pragma unroll
    for (int v = 0; v < 8; ++v)
        out[((size_t)(mt * 16 + lg * 8 + v)) * N + nt * 16 + ln] = acc[v];
}

// ---------------- attention conv as 9 shifted GEMMs over padded input ----------------
// x_em[b*240+pos][e] = bias[e] + sum_tap cfpad_shift @ wtap^T ; 960 x 32 tiles
__global__ void attconv_gemm_kernel(const bf16_t* __restrict__ cfpad,
                                    const bf16_t* __restrict__ wtap,
                                    const float* __restrict__ bias, float* __restrict__ xem) {
    int wid = blockIdx.x * (blockDim.x >> 5) + (threadIdx.x >> 5);
    int mt = wid % 960, nt = wid / 960;
    int l = threadIdx.x & 31, ln = l & 15, lg = l >> 4;
    int m = mt * 16 + ln;
    int b = m / HW, pos = m % HW;
    int y = pos / MW, x = pos % MW;
    v8f acc;
    float bv = bias[nt * 16 + ln];
    #pragma unroll
    for (int v = 0; v < 8; ++v) acc[v] = bv;
    for (int tap = 0; tap < 9; ++tap) {
        int dy = tap / 3, dx = tap % 3;
        const bf16_t* arow = cfpad + ((size_t)((b * 8 + y + dy) * 42 + (x + dx))) * FF;
        const bf16_t* w = wtap + (size_t)tap * EE * FF;
        for (int kt = 0; kt < FF / 32; ++kt)
            acc = wmma_bf16(load_a_frag(arow, kt * 32),
                            load_b_frag(w, FF, nt * 16, kt * 32), acc);
    }
    #pragma unroll
    for (int v = 0; v < 8; ++v)
        xem[((size_t)(mt * 16 + lg * 8 + v)) * EE + nt * 16 + ln] = acc[v];
}

// ---------------- attention score / softmax / pooling ----------------
__global__ void escore_kernel(const float* __restrict__ xem, const float* __restrict__ gem,
                              const float* __restrict__ wv, const float* __restrict__ bv,
                              float* __restrict__ ebuf) {
    int wid = blockIdx.x * (blockDim.x >> 5) + (threadIdx.x >> 5);  // row in [0,15360)
    int l = threadIdx.x & 31;
    int b = wid / HW;
    const float* xr = xem + (size_t)wid * EE;
    const float* gr = gem + (size_t)b * EE;
    float s = 0.f;
    for (int e = l; e < EE; e += 32) s += tanhf(xr[e] + gr[e]) * wv[e];
    #pragma unroll
    for (int off = 16; off; off >>= 1) s += __shfl_xor(s, off, 32);
    if (l == 0) ebuf[wid] = s + bv[0];
}
__global__ void softmax240_kernel(float* __restrict__ ebuf) {
    __shared__ float red[256];
    __shared__ float mv, sv;
    int b = blockIdx.x, t = threadIdx.x;
    float* row = ebuf + (size_t)b * HW;
    float x = (t < HW) ? row[t] : -INFINITY;
    red[t] = x; __syncthreads();
    for (int s = 128; s; s >>= 1) { if (t < s) red[t] = fmaxf(red[t], red[t + s]); __syncthreads(); }
    if (t == 0) mv = red[0];
    __syncthreads();
    float ex = (t < HW) ? __expf(x - mv) : 0.f;
    red[t] = ex; __syncthreads();
    for (int s = 128; s; s >>= 1) { if (t < s) red[t] += red[t + s]; __syncthreads(); }
    if (t == 0) sv = red[0];
    __syncthreads();
    if (t < HW) row[t] = ex / sv;
}
__global__ void attpool_kernel(const float* __restrict__ alpha, const bf16_t* __restrict__ cfpad,
                               bf16_t* __restrict__ attbf) {
    __shared__ float al[HW];
    int b = blockIdx.x, t = threadIdx.x;  // t = feature 0..511
    if (t < HW) al[t] = alpha[(size_t)b * HW + t];
    __syncthreads();
    float s = 0.f;
    int w = 0;
    for (int y = 0; y < MH; ++y)
        for (int x = 0; x < MW; ++x, ++w)
            s += al[w] * (float)cfpad[((size_t)(b * 8 + y + 1) * 42 + (x + 1)) * FF + t];
    attbf[(size_t)b * FF + t] = (bf16_t)s;
}

// ---------------- final masked-NLL loss over stored logits [T][B][V2] ----------------
__global__ void loss_kernel(const float* __restrict__ logits, const int* __restrict__ tgt,
                            float* __restrict__ out) {
    __shared__ float snll[256], smask[256];
    int t = threadIdx.x;
    float an = 0.f, am = 0.f;
    for (int p = t; p < TT * BB; p += 256) {
        int tt = p >> 6, b = p & 63;
        int lt = (tt == 0 || tt == TT - 1) ? 0 : tgt[b * 30 + tt - 1];
        int j = TT - 1;
        for (int q = 1; q < TT; ++q) {
            int v = (q == TT - 1) ? 0 : tgt[b * 30 + q - 1];
            if (v == 0) { j = q; break; }
        }
        int eff = (tt == j) ? ENDTOK : lt;
        if (eff != 0) {
            const float* row = logits + ((size_t)tt * BB + b) * V2;
            float mx = row[0];
            for (int v = 1; v < V2; ++v) mx = fmaxf(mx, row[v]);
            float se = 0.f;
            for (int v = 0; v < V2; ++v) se += __expf(row[v] - mx);
            an += -(row[eff] - (__logf(se) + mx));
            am += 1.f;
        }
    }
    snll[t] = an; smask[t] = am; __syncthreads();
    for (int s = 128; s; s >>= 1) {
        if (t < s) { snll[t] += snll[t + s]; smask[t] += smask[t + s]; }
        __syncthreads();
    }
    if (t == 0) out[0] = snll[0] / smask[0];
}

// ---------------- host orchestration ----------------
extern "C" void kernel_launch(void* const* d_in, const int* in_sizes, int n_in,
                              void* d_out, int out_size, void* d_ws, size_t ws_size,
                              hipStream_t stream) {
    (void)in_sizes; (void)n_in; (void)out_size;
    const float* conv_f    = (const float*)d_in[0];
    const int*   target    = (const int*)d_in[1];
    const float* w_ih_en   = (const float*)d_in[2];
    const float* w_hh_en   = (const float*)d_in[3];
    const float* b_ih_en   = (const float*)d_in[4];
    const float* b_hh_en   = (const float*)d_in[5];
    const float* embed_w   = (const float*)d_in[6];
    const float* w_ih_de   = (const float*)d_in[7];
    const float* w_hh_de   = (const float*)d_in[8];
    const float* b_ih_de   = (const float*)d_in[9];
    const float* b_hh_de   = (const float*)d_in[10];
    const float* att_wh    = (const float*)d_in[11];
    const float* att_bh    = (const float*)d_in[12];
    const float* att_cw    = (const float*)d_in[13];
    const float* att_cb    = (const float*)d_in[14];
    const float* att_wv    = (const float*)d_in[15];
    const float* att_bv    = (const float*)d_in[16];
    const float* out_w     = (const float*)d_in[17];
    const float* out_b     = (const float*)d_in[18];
    float* out = (float*)d_out;

    char* base = (char*)d_ws;
    size_t off = 0;
    auto alloc = [&](size_t bytes) -> char* {
        char* p = base + off;
        off = (off + bytes + 255) & ~(size_t)255;
        return p;
    };
    const size_t WSZ = (size_t)2 * 2048 * 512;  // per-direction LSTM weight elems
    bf16_t* wih_en_bf = (bf16_t*)alloc(WSZ * 2);
    bf16_t* whh_en_bf = (bf16_t*)alloc(WSZ * 2);
    bf16_t* wih_de_bf = (bf16_t*)alloc(WSZ * 2);
    bf16_t* whh_de_bf = (bf16_t*)alloc(WSZ * 2);
    bf16_t* att_wh_bf = (bf16_t*)alloc((size_t)EE * HH * 2);
    bf16_t* out_w_bf  = (bf16_t*)alloc((size_t)V2 * 1024 * 2);
    bf16_t* wtap_bf   = (bf16_t*)alloc((size_t)9 * EE * FF * 2);
    bf16_t* xseq_bf   = (bf16_t*)alloc((size_t)MW * BB * FF * 2);
    bf16_t* cfpad_bf  = (bf16_t*)alloc((size_t)BB * 8 * 42 * FF * 2);
    float*  x_em      = (float*)alloc((size_t)MROWS * EE * 4);
    bf16_t* xs_bf     = (bf16_t*)alloc((size_t)TT * BB * HH * 2);
    float*  c_all     = (float*)alloc((size_t)4 * BB * HH * 4);     // en0,en1,de0,de1
    bf16_t* h_all     = (bf16_t*)alloc((size_t)8 * BB * HH * 2);    // 4 states x 2 parity
    float*  g_em      = (float*)alloc((size_t)BB * EE * 4);
    float*  e_buf     = (float*)alloc((size_t)BB * HW * 4);
    bf16_t* att_b16   = (bf16_t*)alloc((size_t)BB * FF * 2);
    float*  logits    = (float*)alloc((size_t)TT * BB * V2 * 4);
    if (off > ws_size) return;  // insufficient scratch: leave output poisoned

    const size_t SB = (size_t)BB * HH;  // 32768 elems per state matrix
    float*  c_en0 = c_all;            float* c_en1 = c_all + SB;
    float*  c_de0 = c_all + 2 * SB;   float* c_de1 = c_all + 3 * SB;
    bf16_t* h_en0[2] = { h_all,          h_all + SB };
    bf16_t* h_en1[2] = { h_all + 2*SB,   h_all + 3*SB };
    bf16_t* h_de0[2] = { h_all + 4*SB,   h_all + 5*SB };
    bf16_t* h_de1[2] = { h_all + 6*SB,   h_all + 7*SB };

    auto fillz = [&](void* p, size_t bytes) {
        int nw = (int)(bytes / 4);
        fillz_kernel<<<(nw + 255) / 256, 256, 0, stream>>>((unsigned int*)p, nw);
    };
    auto cvt = [&](const float* s, bf16_t* d, size_t n) {
        f2bf_kernel<<<((int)n + 255) / 256, 256, 0, stream>>>(s, d, (int)n);
    };

    // --- init & preprocessing ---
    fillz(c_all, 4 * SB * 4);
    fillz(h_all, 8 * SB * 2);
    fillz(cfpad_bf, (size_t)BB * 8 * 42 * FF * 2);
    cvt(w_ih_en, wih_en_bf, WSZ);  cvt(w_hh_en, whh_en_bf, WSZ);
    cvt(w_ih_de, wih_de_bf, WSZ);  cvt(w_hh_de, whh_de_bf, WSZ);
    cvt(att_wh, att_wh_bf, (size_t)EE * HH);
    cvt(out_w, out_w_bf, (size_t)V2 * 1024);
    convw_reorg_kernel<<<(9 * EE * FF + 255) / 256, 256, 0, stream>>>(att_cw, wtap_bf);
    maxpool_kernel<<<(MW * BB * FF + 255) / 256, 256, 0, stream>>>(conv_f, xseq_bf);
    padfill_kernel<<<(BB * MH * MW * FF + 255) / 256, 256, 0, stream>>>(conv_f, cfpad_bf);
    // attention conv embedding (one-time): 960x32 tiles, 8 waves/block
    attconv_gemm_kernel<<<(960 * 32) / 8, 256, 0, stream>>>(cfpad_bf, wtap_bf, att_cb, x_em);
    xsembed_kernel<<<((TT - 1) * BB * HH + 255) / 256, 256, 0, stream>>>(embed_w, target, xs_bf);

    // --- encoder: 40 steps x 2 layers ---
    int p0 = 0, p1 = 0;
    for (int t = 0; t < MW; ++t) {
        lstm_step_kernel<<<32, 128, 0, stream>>>(
            xseq_bf + (size_t)t * BB * FF, h_en0[p0], wih_en_bf, whh_en_bf,
            b_ih_en, b_hh_en, c_en0, h_en0[p0 ^ 1]);
        p0 ^= 1;
        lstm_step_kernel<<<32, 128, 0, stream>>>(
            h_en0[p0], h_en1[p1], wih_en_bf + WSZ / 2, whh_en_bf + WSZ / 2,
            b_ih_en + 2048, b_hh_en + 2048, c_en1, h_en1[p1 ^ 1]);
        p1 ^= 1;
    }
    // holistic -> xs[0]
    hipMemcpyAsync(xs_bf, h_en1[p1], SB * sizeof(bf16_t), hipMemcpyDeviceToDevice, stream);

    // --- decoder: 32 steps ---
    int q0 = 0, q1 = 0;
    for (int t = 0; t < TT; ++t) {
        lstm_step_kernel<<<32, 128, 0, stream>>>(
            xs_bf + (size_t)t * BB * HH, h_de0[q0], wih_de_bf, whh_de_bf,
            b_ih_de, b_hh_de, c_de0, h_de0[q0 ^ 1]);
        q0 ^= 1;
        lstm_step_kernel<<<32, 128, 0, stream>>>(
            h_de0[q0], h_de1[q1], wih_de_bf + WSZ / 2, whh_de_bf + WSZ / 2,
            b_ih_de + 2048, b_hh_de + 2048, c_de1, h_de1[q1 ^ 1]);
        q1 ^= 1;
        // g_em = hlast @ att_wh^T + att_bh   (4x32 tiles)
        gemm_bias_kernel<<<32, 128, 0, stream>>>(h_de1[q1], att_wh_bf, HH, HH,
                                                 nullptr, nullptr, 0, 0,
                                                 att_bh, g_em, EE, 32);
        escore_kernel<<<MROWS / 8, 256, 0, stream>>>(x_em, g_em, att_wv, att_bv, e_buf);
        softmax240_kernel<<<BB, 256, 0, stream>>>(e_buf);
        attpool_kernel<<<BB, 512, 0, stream>>>(e_buf, cfpad_bf, att_b16);
        // logits = [hlast, att] @ out_w^T + out_b   (4x6 tiles)
        gemm_bias_kernel<<<6, 128, 0, stream>>>(h_de1[q1], out_w_bf, HH, 1024,
                                                att_b16, out_w_bf + 512, FF, 1024,
                                                out_b, logits + (size_t)t * BB * V2, V2, 6);
    }
    loss_kernel<<<1, 256, 0, stream>>>(logits, target, out);
}
